// InvariantBlock_66571993088802
// MI455X (gfx1250) — matrix operand order repaired and checked by reading, not compiled, not run on hardware
//
#include <hip/hip_runtime.h>
#include <hip/hip_bf16.h>
#include <math.h>

// ---------------- problem dims ----------------
#define B_    2
#define L_    512
#define LP1   513
#define D_    512
#define H_    8
#define HD_   64
#define FF_   2048
#define RFFD  32
#define RBH   64

#define MROWS 1026          // B_*LP1
#define MPAD  1040          // 65 tiles of 16
#define MT_   65
#define LPAD  544           // 34 tiles of 16 (key/seq padding, covers K=32 V reads)
#define KT_   33            // 33 key tiles of 16 cover 513..528
#define PAIRS 263169        // 513*513
#define PT_   16449         // ceil(PAIRS/16)

typedef __attribute__((ext_vector_type(16))) _Float16 v16h;
typedef __attribute__((ext_vector_type(8)))  _Float16 v8h;
typedef __attribute__((ext_vector_type(8)))  float    v8f;

// ---------------- WMMA helpers ----------------
__device__ __forceinline__ v16h mk16(v8h lo, v8h hi) {
    v16h r;
#pragma unroll
    for (int i = 0; i < 8; i++) { r[i] = lo[i]; r[i + 8] = hi[i]; }
    return r;
}

// A operand (16x32, M x K, f16), row-major source with row stride ldk halfs.
// Lane l: M=l&15, K = {kb..kb+7, kb+16..kb+23}, kb = (l>=16)*8
__device__ __forceinline__ v16h loadA16(const _Float16* base, int ldk, int lane) {
    int row = lane & 15;
    int kb  = (lane >> 4) ? 8 : 0;
    const _Float16* p = base + row * ldk + kb;
    return mk16(*(const v8h*)p, *(const v8h*)(p + 16));
}

// B operand (32x16, K x N, f16) from weight stored as [N][K] row-major (stride ldk).
// Lane l: N = l&15, K = {kb..kb+7, kb+16..kb+23}, kb = (l>=16)*8
__device__ __forceinline__ v16h loadB16(const _Float16* base, int ldk, int lane) {
    int col = lane & 15;
    int kb  = (lane >> 4) ? 8 : 0;
    const _Float16* p = base + col * ldk + kb;
    return mk16(*(const v8h*)p, *(const v8h*)(p + 16));
}

__device__ __forceinline__ v8f wmma_f16(v16h a, v16h b, v8f c) {
    return __builtin_amdgcn_wmma_f32_16x16x32_f16(false, a, false, b, (short)0, c, false, false);
}

__device__ __forceinline__ v8f zero8() {
    v8f z = {0.f, 0.f, 0.f, 0.f, 0.f, 0.f, 0.f, 0.f};
    return z;
}

// C[m][n] = sum_k A[m][k] * W[n][k]; one wave -> 16 x 64 strip (4 accumulators).
// Software-pipelined: k-step i+1's loads are issued before k-step i's WMMAs so
// the per-WMMA s_wait_loadcnt becomes partial and load latency hides in XDL time.
template <int KTOT>
__device__ __forceinline__ void gemm_row64(const _Float16* A, const _Float16* W,
                                           int lda, int ldw, int m0, int n0,
                                           v8f acc[4], int lane) {
    const _Float16* Abase = A + (size_t)m0 * lda;
    const _Float16* Wbase = W + (size_t)n0 * ldw;
    v16h a0 = loadA16(Abase, lda, lane);
    v16h b0[4];
#pragma unroll
    for (int nt = 0; nt < 4; nt++) b0[nt] = loadB16(Wbase + (size_t)nt * 16 * ldw, ldw, lane);
#pragma unroll 4
    for (int k0 = 32; k0 < KTOT; k0 += 32) {
        // issue next k-step's loads first
        v16h a1 = loadA16(Abase + k0, lda, lane);
        v16h b1[4];
#pragma unroll
        for (int nt = 0; nt < 4; nt++)
            b1[nt] = loadB16(Wbase + (size_t)nt * 16 * ldw + k0, ldw, lane);
        // consume previous k-step
#pragma unroll
        for (int nt = 0; nt < 4; nt++) acc[nt] = wmma_f16(a0, b0[nt], acc[nt]);
        a0 = a1;
#pragma unroll
        for (int nt = 0; nt < 4; nt++) b0[nt] = b1[nt];
    }
#pragma unroll
    for (int nt = 0; nt < 4; nt++) acc[nt] = wmma_f16(a0, b0[nt], acc[nt]);
}

__device__ __forceinline__ float gelu_exact(float x) {
    return 0.5f * x * (1.0f + erff(x * 0.70710678118654752f));
}

// ---------------- elementwise helpers ----------------
__global__ void __launch_bounds__(256) f32_to_f16_kernel(const float* in, _Float16* out, int n) {
    int i = blockIdx.x * 256 + threadIdx.x;
    if (i < n) out[i] = (_Float16)in[i];
}

// one wave per row of 512; out = LayerNorm(x)*g + b, stored f16
__global__ void __launch_bounds__(128) ln_f16_kernel(const float* x, const float* g,
                                                     const float* bta, _Float16* out, int rows) {
    int wid  = blockIdx.x * 4 + (threadIdx.x >> 5);
    int lane = threadIdx.x & 31;
    if (wid >= rows) return;
    const float* xr = x + (size_t)wid * D_;
    float vals[16];
    float s = 0.f, s2 = 0.f;
#pragma unroll
    for (int i = 0; i < 16; i++) {
        float v = xr[lane + 32 * i];
        vals[i] = v; s += v; s2 += v * v;
    }
#pragma unroll
    for (int off = 16; off >= 1; off >>= 1) {
        s  += __shfl_xor(s,  off, 32);
        s2 += __shfl_xor(s2, off, 32);
    }
    float mu  = s  * (1.0f / D_);
    float var = s2 * (1.0f / D_) - mu * mu;
    float inv = rsqrtf(var + 1e-5f);
#pragma unroll
    for (int i = 0; i < 16; i++) {
        int c = lane + 32 * i;
        out[(size_t)wid * D_ + c] = (_Float16)((vals[i] - mu) * inv * g[c] + bta[c]);
    }
}

// ---------------- QKV projection (WMMA GEMM, scattered epilogue) ----------------
__global__ void __launch_bounds__(128) qkv_gemm_kernel(const _Float16* zh, const _Float16* wh_in,
                                                       const float* in_b, _Float16* qh,
                                                       _Float16* kh, _Float16* vt) {
    int wid  = blockIdx.x * 4 + (threadIdx.x >> 5);
    int lane = threadIdx.x & 31;
    if (wid >= MT_ * 24) return;
    int mt = wid / 24, ng = wid % 24;
    int m0 = mt * 16, n0 = ng * 64;
    v8f acc[4] = {zero8(), zero8(), zero8(), zero8()};
    gemm_row64<512>(zh, wh_in, 512, 512, m0, n0, acc, lane);
    int hg = (lane >> 4) << 3;
#pragma unroll
    for (int nt = 0; nt < 4; nt++) {
        int n = n0 + nt * 16 + (lane & 15);
        float bn  = in_b[n];
        int which = n >> 9;          // 0:q 1:k 2:v
        int h     = (n >> 6) & 7;
        int d     = n & 63;
#pragma unroll
        for (int r = 0; r < 8; r++) {
            int m = m0 + r + hg;
            if (m < MROWS) {
                int b = m / LP1, t = m % LP1;
                int bh = b * H_ + h;
                _Float16 v = (_Float16)(acc[nt][r] + bn);
                if (which == 0)      qh[((size_t)bh * LPAD + t) * HD_ + d] = v;
                else if (which == 1) kh[((size_t)bh * LPAD + t) * HD_ + d] = v;
                else                 vt[((size_t)bh * HD_ + d) * LPAD + t] = v;
            }
        }
    }
}

// ---------------- relative-bias MLP (WMMA), 16 pairs per wave ----------------
__global__ void __launch_bounds__(128) bias_mlp_kernel(const float* s_norm, const float* rff,
                                                       const float* w1, const float* b1,
                                                       const float* w2, const float* b2,
                                                       float* bias_out) {
    __shared__ __align__(16) _Float16 hid_lds[4][16][RBH];
    int wv   = threadIdx.x >> 5;
    int wid  = blockIdx.x * 4 + wv;
    int lane = threadIdx.x & 31;
    if (wid >= B_ * PT_) return;
    int b  = wid / PT_;
    int pt = wid % PT_;

    // per-lane pair delta (row of A); rows beyond PAIRS -> ds=0 (harmless)
    int p = pt * 16 + (lane & 15);
    float ds = 0.f;
    if (p < PAIRS) {
        int i = p / LP1, j = p % LP1;
        float si = (i == 0) ? 0.f : s_norm[b * L_ + i - 1];
        float sj = (j == 0) ? 0.f : s_norm[b * L_ + j - 1];
        ds = si - sj;
    }
    // A = sin features in WMMA A layout (K=32, one step)
    int kbA = (lane >> 4) ? 8 : 0;
    v16h a;
#pragma unroll
    for (int e = 0; e < 16; e++) {
        int k = ((e < 8) ? e : e + 8) + kbA;
        a[e] = (_Float16)__sinf(ds * rff[k]);
    }
    // GEMM1: hid(16x64) = feat(16x32) @ w1^T  (w1: [64][32] f32)
    int coln = lane & 15;
    int kbB  = (lane >> 4) ? 8 : 0;
    v8f acc[4] = {zero8(), zero8(), zero8(), zero8()};
#pragma unroll
    for (int nt = 0; nt < 4; nt++) {
        const float* wr = w1 + (nt * 16 + coln) * RFFD + kbB;
        v16h bv;
#pragma unroll
        for (int e = 0; e < 16; e++) bv[e] = (_Float16)wr[(e < 8) ? e : e + 8];
        acc[nt] = wmma_f16(a, bv, acc[nt]);
    }
    int hg = (lane >> 4) << 3;
    // +b1, GELU, stage to LDS for re-layout
#pragma unroll
    for (int nt = 0; nt < 4; nt++) {
        int n = nt * 16 + coln;
        float bb = b1[n];
#pragma unroll
        for (int r = 0; r < 8; r++) {
            hid_lds[wv][r + hg][n] = (_Float16)gelu_exact(acc[nt][r] + bb);
        }
    }
    // GEMM2: out(16x16, 8 valid cols) = hid(16x64) @ w2^T (w2: [8][64] f32)
    v8f acc2 = zero8();
#pragma unroll
    for (int ks = 0; ks < RBH; ks += 32) {
        const _Float16* pr = &hid_lds[wv][lane & 15][ks + kbA];
        v16h a2 = mk16(*(const v8h*)pr, *(const v8h*)(pr + 16));
        v16h b2v;
        if (coln < 8) {
            const float* wr = w2 + coln * RBH + ks + kbB;
#pragma unroll
            for (int e = 0; e < 16; e++) b2v[e] = (_Float16)wr[(e < 8) ? e : e + 8];
        } else {
#pragma unroll
            for (int e = 0; e < 16; e++) b2v[e] = (_Float16)0.f;
        }
        acc2 = wmma_f16(a2, b2v, acc2);
    }
    if (coln < 8) {
        float bb2 = b2[coln];
        size_t basep = ((size_t)b * H_ + coln) * PAIRS;
#pragma unroll
        for (int r = 0; r < 8; r++) {
            int pp = pt * 16 + r + hg;
            if (pp < PAIRS) bias_out[basep + pp] = acc2[r] + bb2;
        }
    }
}

// ---------------- fused flash attention (one wave = one (b,h,qtile)) ----------------
__global__ void __launch_bounds__(128) attn_kernel(const _Float16* qh, const _Float16* kh,
                                                   const _Float16* vt, const float* bias,
                                                   _Float16* ah) {
    __shared__ __align__(16) _Float16 p_lds[4][16][32];
    int wv   = threadIdx.x >> 5;
    int wid  = blockIdx.x * 4 + wv;
    int lane = threadIdx.x & 31;
    if (wid >= B_ * H_ * KT_) return;
    int b  = wid / (H_ * KT_);
    int h  = (wid / KT_) % H_;
    int qt = wid % KT_;
    int bh = b * H_ + h;
    int kbA  = (lane >> 4) ? 8 : 0;
    int kbB  = (lane >> 4) ? 8 : 0;
    int coln = lane & 15;
    int hg   = (lane >> 4) << 3;

    // Q tile (16x64) in A layout, two K-steps of 32
    v16h aq[2];
    {
        const _Float16* qrow = qh + ((size_t)bh * LPAD + qt * 16 + (lane & 15)) * HD_;
#pragma unroll
        for (int s = 0; s < 2; s++) {
            const _Float16* pq = qrow + s * 32 + kbA;
            aq[s] = mk16(*(const v8h*)pq, *(const v8h*)(pq + 16));
        }
    }
    // zero the whole P staging tile once (upper 16 K cols stay 0 forever)
    {
        _Float16* pz = &p_lds[wv][0][0];
#pragma unroll
        for (int i = 0; i < 16; i++) pz[lane * 16 + i] = (_Float16)0.f;
    }
    float mrow[8], lrow[8];
#pragma unroll
    for (int r = 0; r < 8; r++) { mrow[r] = -1e30f; lrow[r] = 0.f; }
    v8f accO[4] = {zero8(), zero8(), zero8(), zero8()};

    for (int kt = 0; kt < KT_; kt++) {
        int key0 = kt * 16;
        int n_   = key0 + coln;

        // --- issue ALL independent loads for this tile up-front ---
        // K tile B-operands (for S), V tile B-operands (for O), bias row values.
        v16h bk[2];
        {
            const _Float16* krow = kh + ((size_t)bh * LPAD + key0 + coln) * HD_;
#pragma unroll
            for (int s = 0; s < 2; s++) {
                const _Float16* pk = krow + s * 32 + kbB;
                bk[s] = mk16(*(const v8h*)pk, *(const v8h*)(pk + 16));
            }
        }
        v16h bvv[4];
#pragma unroll
        for (int nt = 0; nt < 4; nt++) {
            int d = nt * 16 + coln;
            const _Float16* vp = vt + ((size_t)bh * HD_ + d) * LPAD + key0 + kbB;
            bvv[nt] = mk16(*(const v8h*)vp, *(const v8h*)(vp + 16));
        }
        float bs[8];
#pragma unroll
        for (int r = 0; r < 8; r++) {
            int m_ = qt * 16 + r + hg;
            bs[r] = (n_ < LP1 && m_ < LP1)
                        ? bias[(size_t)bh * PAIRS + (size_t)m_ * LP1 + n_] : 0.f;
        }

        // S = Q @ K^T  (C layout: row = r+hg of q tile, col = key0+coln)
        v8f sa = zero8();
        sa = wmma_f16(aq[0], bk[0], sa);
        sa = wmma_f16(aq[1], bk[1], sa);

        // online softmax in C layout (V loads still in flight during this VALU work)
#pragma unroll
        for (int r = 0; r < 8; r++) {
            float sv = (n_ < LP1) ? (sa[r] * 0.125f + bs[r]) : -1e30f;
            float mx = sv;
#pragma unroll
            for (int off = 8; off >= 1; off >>= 1) mx = fmaxf(mx, __shfl_xor(mx, off, 32));
            float mnew  = fmaxf(mrow[r], mx);
            float scale = __expf(mrow[r] - mnew);
            float pv    = __expf(sv - mnew);
            float rs    = pv;
#pragma unroll
            for (int off = 8; off >= 1; off >>= 1) rs += __shfl_xor(rs, off, 32);
            lrow[r] = lrow[r] * scale + rs;
            mrow[r] = mnew;
#pragma unroll
            for (int nt = 0; nt < 4; nt++) accO[nt][r] *= scale;
            p_lds[wv][r + hg][coln] = (_Float16)pv;
        }
        // O += P @ V  (P: 16x32 with zero upper K half; V read from d-major vt)
        const _Float16* pr = &p_lds[wv][lane & 15][kbA];
        v16h ap = mk16(*(const v8h*)pr, *(const v8h*)(pr + 16));
#pragma unroll
        for (int nt = 0; nt < 4; nt++) accO[nt] = wmma_f16(ap, bvv[nt], accO[nt]);
    }
    // epilogue: O / l -> ah[(b*513+t)][h*64+d] f16
#pragma unroll
    for (int r = 0; r < 8; r++) {
        int t = qt * 16 + r + hg;
        if (t < LP1) {
            size_t m  = (size_t)b * LP1 + t;
            float inv = 1.0f / lrow[r];
#pragma unroll
            for (int nt = 0; nt < 4; nt++) {
                int d = nt * 16 + coln;
                ah[m * D_ + h * HD_ + d] = (_Float16)(accO[nt][r] * inv);
            }
        }
    }
}

// ---------------- out-projection + residual ----------------
__global__ void __launch_bounds__(128) outproj_gemm_kernel(const _Float16* ah, const _Float16* wh,
                                                           const float* ob, const float* x,
                                                           float* x1) {
    int wid  = blockIdx.x * 4 + (threadIdx.x >> 5);
    int lane = threadIdx.x & 31;
    if (wid >= MT_ * 8) return;
    int m0 = (wid / 8) * 16, n0 = (wid % 8) * 64;
    v8f acc[4] = {zero8(), zero8(), zero8(), zero8()};
    gemm_row64<512>(ah, wh, 512, 512, m0, n0, acc, lane);
    int hg = (lane >> 4) << 3;
#pragma unroll
    for (int nt = 0; nt < 4; nt++) {
        int n = n0 + nt * 16 + (lane & 15);
        float bn = ob[n];
#pragma unroll
        for (int r = 0; r < 8; r++) {
            int m = m0 + r + hg;
            if (m < MROWS) x1[(size_t)m * D_ + n] = acc[nt][r] + bn + x[(size_t)m * D_ + n];
        }
    }
}

// ---------------- FF1 (+GELU) ----------------
__global__ void __launch_bounds__(128) ff1_gemm_kernel(const _Float16* h2h, const _Float16* wh,
                                                       const float* fb, _Float16* ffh) {
    int wid  = blockIdx.x * 4 + (threadIdx.x >> 5);
    int lane = threadIdx.x & 31;
    if (wid >= MT_ * 32) return;
    int m0 = (wid / 32) * 16, n0 = (wid % 32) * 64;
    v8f acc[4] = {zero8(), zero8(), zero8(), zero8()};
    gemm_row64<512>(h2h, wh, 512, 512, m0, n0, acc, lane);
    int hg = (lane >> 4) << 3;
#pragma unroll
    for (int nt = 0; nt < 4; nt++) {
        int n = n0 + nt * 16 + (lane & 15);
        float bn = fb[n];
#pragma unroll
        for (int r = 0; r < 8; r++) {
            int m = m0 + r + hg;
            if (m < MROWS) ffh[(size_t)m * FF_ + n] = (_Float16)gelu_exact(acc[nt][r] + bn);
        }
    }
}

// ---------------- FF2 + residual -> d_out ----------------
__global__ void __launch_bounds__(128) ff2_gemm_kernel(const _Float16* ffh, const _Float16* wh,
                                                       const float* fb, const float* x1,
                                                       float* out) {
    int wid  = blockIdx.x * 4 + (threadIdx.x >> 5);
    int lane = threadIdx.x & 31;
    if (wid >= MT_ * 8) return;
    int m0 = (wid / 8) * 16, n0 = (wid % 8) * 64;
    v8f acc[4] = {zero8(), zero8(), zero8(), zero8()};
    gemm_row64<2048>(ffh, wh, 2048, 2048, m0, n0, acc, lane);
    int hg = (lane >> 4) << 3;
#pragma unroll
    for (int nt = 0; nt < 4; nt++) {
        int n = n0 + nt * 16 + (lane & 15);
        float bn = fb[n];
#pragma unroll
        for (int r = 0; r < 8; r++) {
            int m = m0 + r + hg;
            if (m < MROWS) out[(size_t)m * D_ + n] = acc[nt][r] + bn + x1[(size_t)m * D_ + n];
        }
    }
}

// ---------------- host orchestration ----------------
extern "C" void kernel_launch(void* const* d_in, const int* in_sizes, int n_in,
                              void* d_out, int out_size, void* d_ws, size_t ws_size,
                              hipStream_t stream) {
    const float* x      = (const float*)d_in[0];
    const float* s_norm = (const float*)d_in[1];
    const float* rff    = (const float*)d_in[2];
    const float* rb_w1  = (const float*)d_in[3];
    const float* rb_b1  = (const float*)d_in[4];
    const float* rb_w2  = (const float*)d_in[5];
    const float* rb_b2  = (const float*)d_in[6];
    const float* n1_g   = (const float*)d_in[7];
    const float* n1_b   = (const float*)d_in[8];
    const float* n2_g   = (const float*)d_in[9];
    const float* n2_b   = (const float*)d_in[10];
    const float* in_w   = (const float*)d_in[11];
    const float* in_b   = (const float*)d_in[12];
    const float* out_w  = (const float*)d_in[13];
    const float* out_b  = (const float*)d_in[14];
    const float* ff_w1  = (const float*)d_in[15];
    const float* ff_b1  = (const float*)d_in[16];
    const float* ff_w2  = (const float*)d_in[17];
    const float* ff_b2  = (const float*)d_in[18];
    float* out = (float*)d_out;

    char* ws = (char*)d_ws;
    size_t off = 0;
    auto alloc = [&](size_t bytes) -> char* {
        char* p = ws + off;
        off = (off + bytes + 255) & ~(size_t)255;
        return p;
    };
    _Float16* wh_in  = (_Float16*)alloc((size_t)3 * D_ * D_ * 2);
    _Float16* wh_out = (_Float16*)alloc((size_t)D_ * D_ * 2);
    _Float16* wh_f1  = (_Float16*)alloc((size_t)FF_ * D_ * 2);
    _Float16* wh_f2  = (_Float16*)alloc((size_t)D_ * FF_ * 2);
    char* zbeg = ws + off;
    _Float16* zh  = (_Float16*)alloc((size_t)MPAD * D_ * 2);
    _Float16* qh  = (_Float16*)alloc((size_t)B_ * H_ * LPAD * HD_ * 2);
    _Float16* kh  = (_Float16*)alloc((size_t)B_ * H_ * LPAD * HD_ * 2);
    _Float16* vt  = (_Float16*)alloc((size_t)B_ * H_ * HD_ * LPAD * 2);
    _Float16* ah  = (_Float16*)alloc((size_t)MPAD * D_ * 2);
    _Float16* h2h = (_Float16*)alloc((size_t)MPAD * D_ * 2);
    _Float16* ffh = (_Float16*)alloc((size_t)MPAD * FF_ * 2);
    char* zend = ws + off;
    float* x1    = (float*)alloc((size_t)MROWS * D_ * 4);
    float* biasb = (float*)alloc((size_t)B_ * H_ * PAIRS * 4);

    // zero-fill all padded f16 staging (pad rows/cols must be exact zeros)
    hipMemsetAsync(zbeg, 0, (size_t)(zend - zbeg), stream);

    // weight f32 -> f16
    f32_to_f16_kernel<<<(3 * D_ * D_ + 255) / 256, 256, 0, stream>>>(in_w, wh_in, 3 * D_ * D_);
    f32_to_f16_kernel<<<(D_ * D_ + 255) / 256, 256, 0, stream>>>(out_w, wh_out, D_ * D_);
    f32_to_f16_kernel<<<(FF_ * D_ + 255) / 256, 256, 0, stream>>>(ff_w1, wh_f1, FF_ * D_);
    f32_to_f16_kernel<<<(D_ * FF_ + 255) / 256, 256, 0, stream>>>(ff_w2, wh_f2, D_ * FF_);

    // LN1 -> zh (f16)
    ln_f16_kernel<<<(MROWS + 3) / 4, 128, 0, stream>>>(x, n1_g, n1_b, zh, MROWS);

    // QKV projection -> q/k head-major, v transposed (d-major)
    qkv_gemm_kernel<<<(MT_ * 24 + 3) / 4, 128, 0, stream>>>(zh, wh_in, in_b, qh, kh, vt);

    // relative-position bias MLP (WMMA) -> (B,H,513,513)
    bias_mlp_kernel<<<(B_ * PT_ + 3) / 4, 128, 0, stream>>>(s_norm, rff, rb_w1, rb_b1,
                                                            rb_w2, rb_b2, biasb);

    // fused flash attention
    attn_kernel<<<(B_ * H_ * KT_ + 3) / 4, 128, 0, stream>>>(qh, kh, vt, biasb, ah);

    // out-projection + residual -> x1 (f32)
    outproj_gemm_kernel<<<(MT_ * 8 + 3) / 4, 128, 0, stream>>>(ah, wh_out, out_b, x, x1);

    // LN2 -> h2h (f16)
    ln_f16_kernel<<<(MROWS + 3) / 4, 128, 0, stream>>>(x1, n2_g, n2_b, h2h, MROWS);

    // FF1 (+GELU) and FF2 (+residual) -> d_out
    ff1_gemm_kernel<<<(MT_ * 32 + 3) / 4, 128, 0, stream>>>(h2h, wh_f1, ff_b1, ffh);
    ff2_gemm_kernel<<<(MT_ * 8 + 3) / 4, 128, 0, stream>>>(ffh, wh_f2, ff_b2, x1, out);
}
